// Unrolled_ConvLstm_84301618086335
// MI455X (gfx1250) — compile-verified, matrix operand order checked by reading, not run on hardware
//
#include <hip/hip_runtime.h>

typedef __attribute__((ext_vector_type(16))) _Float16 v16h;
typedef __attribute__((ext_vector_type(8)))  _Float16 h8;
typedef __attribute__((ext_vector_type(8)))  float    v8f;

#define Bb    16
#define CIN   8
#define SEQ   32
#define Hh    128
#define Ww    128
#define COUT  8
#define CTOT  16      // Cin + Cout
#define NG    32      // 4*Cout gate channels
#define HW    (Hh*Ww)
#define HP    130     // padded h rows (1-halo)
#define WP    130     // padded h cols (1-halo)
#define ROWS  4       // output rows per workgroup
#define SROWS 6       // staged input rows (ROWS + 2 halo)
// padded, channel-interleaved f16 h buffer: [b][yp][xp][8], bytes:
#define HBYTES ((size_t)Bb * HP * WP * COUT * 2)   // 4,326,400
#define CBYTES ((size_t)Bb * COUT * HW * 4)        // 8,388,608

// Branch-free activations on native v_exp_f32 / v_rcp_f32.
__device__ __forceinline__ float fast_sigmoid(float z) {
  return __builtin_amdgcn_rcpf(1.0f + __expf(-z));
}
__device__ __forceinline__ float fast_tanh(float x) {
  float e = __expf(2.0f * x);
  return (e - 1.0f) * __builtin_amdgcn_rcpf(e + 1.0f);
}

// ---------------------------------------------------------------------------
// Pack conv weights (fp32 [32][16][3][3]) into the exact per-lane f16 B-matrix
// fragment layout for V_WMMA_F32_16X16X32_F16:
//   [ntile(2)][kiter(5)][lane(32)][elem(16)] halves.
// K = (tap, channel): kiter covers taps {2k, 2k+1}; K<16 -> tap 2k, K>=16 ->
// tap 2k+1 (kiter 4 upper half zero-padded, 144 -> 160). ISA 16-bit layout:
// lanes 0-15 hold K {0..7,16..23}, lanes 16-31 hold K {8..15,24..31}.
// ---------------------------------------------------------------------------
__global__ void convlstm_pack_w(const float* __restrict__ Wc,
                                _Float16* __restrict__ wpack) {
  int i = blockIdx.x * blockDim.x + threadIdx.x;
  if (i >= 5120) return;
  int nt   = i / 2560;
  int rem  = i % 2560;
  int ki   = rem / 512;
  int r2   = rem % 512;
  int lane = r2 / 16;
  int e    = r2 % 16;
  int hs   = lane >> 4;
  int n    = nt * 16 + (lane & 15);
  int Kv   = (e < 8) ? (hs * 8 + e) : (16 + hs * 8 + (e - 8));
  int tp   = ki * 2 + (Kv >= 16 ? 1 : 0);
  float v  = 0.0f;
  if (tp <= 8) {
    int c  = Kv & 15;
    int ky = tp / 3, kx = tp % 3;
    v = Wc[((n * CTOT + c) * 3 + ky) * 3 + kx];
  }
  wpack[i] = (_Float16)v;
}

__global__ void convlstm_zero_f32(float* __restrict__ p, int n) {
  for (int i = blockIdx.x * blockDim.x + threadIdx.x; i < n;
       i += gridDim.x * blockDim.x)
    p[i] = 0.0f;
}

// ---------------------------------------------------------------------------
// One time step: z = conv3x3([x_t || h_{t-1}]) + b ; LSTM gate update.
// Grid: (y-block=32, b=16), 256 threads = 8 waves. Each WG produces 4 output
// rows from a 6-row staged window (2x staging reuse). Wave w owns x-tile
// [16w,16w+16) and runs 4 row-tiles of 10 v_wmma_f32_16x16x32_f16 each.
//
// Staging: h rows arrive via GLOBAL_LOAD_ASYNC_TO_LDS_B128 (f16 interleaved,
// pre-zeroed halo ring -> no bounds checks/conversion, ASYNCcnt tracked);
// X rows are converted fp32->f16 by VALU while asyncs are in flight. LDS is
// channel-innermost so each WMMA A-fragment half is one 16B ds_load_b128.
// The per-wave z-exchange region is reused across row-tiles with no barrier:
// same-wave LDS ops are in-order (DScnt).
// ---------------------------------------------------------------------------
__global__ __launch_bounds__(256)
void convlstm_step(const float* __restrict__ X,
                   const h8*      __restrict__ wpack,
                   const float*   __restrict__ bc,
                   const _Float16* __restrict__ h_in,   // [b][yp130][xp130][8]
                   _Float16*       __restrict__ h_out,  // same layout
                   float* __restrict__ c_buf,           // [b][co][y][x]
                   float* __restrict__ out,
                   int t) {
  // 6 rows x 130 x-positions x 16 channels f16, channel innermost (24,960 B)
  __shared__ __attribute__((aligned(16))) _Float16 lds_in[SROWS * 130 * 16];
  __shared__ __attribute__((aligned(16))) float    zbuf[8 * NG * 16];

  const int b  = blockIdx.y;
  const int y0 = blockIdx.x * ROWS;

  // ---- async-stage h window: 6*130 chunks of 16B (8 ch at one x) ----
  {
    const unsigned lds_base = (unsigned)(size_t)(&lds_in[0]);
    for (int k = threadIdx.x; k < SROWS * 130; k += 256) {
      int r  = k / 130;         // staged row: actual y0-1+r == padded y0+r
      int xh = k % 130;         // padded col
      unsigned goff = ((unsigned)((b * HP + (y0 + r)) * WP + xh)) * 16u;
      unsigned ldsa = lds_base + ((unsigned)((r * 130 + xh) * 16 + 8)) * 2u;
      asm volatile("global_load_async_to_lds_b128 %0, %1, %2 offset:0"
                   :: "v"(ldsa), "v"(goff), "s"(h_in)
                   : "memory");
    }
  }

  // ---- stage X window (fp32 -> f16) while asyncs are in flight ----
  for (int i = threadIdx.x; i < CIN * SROWS * 130; i += 256) {
    int c  = i / (SROWS * 130);          // channel 0..7
    int r  = (i % (SROWS * 130)) / 130;  // staged row 0..5
    int xh = i % 130;                    // x+1 (halo)
    int yy = y0 + r - 1;
    int xx = xh - 1;
    float v = 0.0f;
    if ((unsigned)yy < (unsigned)Hh && (unsigned)xx < (unsigned)Ww)
      v = X[(((b * CIN + c) * SEQ + t) * Hh + yy) * Ww + xx];
    lds_in[(r * 130 + xh) * 16 + c] = (_Float16)v;
  }

  // warm next time-step's X slice in GL2: 8ch x 4rows x 8 chunks = 256 threads
  if (t + 1 < SEQ) {
    int c  = threadIdx.x >> 5;
    int j  = (threadIdx.x >> 3) & 3;
    int xq = (threadIdx.x & 7) * 16;
    __builtin_prefetch(
        &X[(((b * CIN + c) * SEQ + (t + 1)) * Hh + (y0 + j)) * Ww + xq], 0, 1);
  }

  const int lane = threadIdx.x & 31;
  const int wv   = threadIdx.x >> 5;
  const int x0   = wv * 16;
  const int m    = lane & 15;
  const int hs   = lane >> 4;

  // ---- preload packed weight (B) fragments: 2 N-tiles x 5 K-iters ----
  v16h bf[2][5];
#pragma unroll
  for (int nt = 0; nt < 2; ++nt) {
#pragma unroll
    for (int ki = 0; ki < 5; ++ki) {
      int base = ((nt * 5 + ki) * 32 + lane) * 2;
      h8 lo = wpack[base + 0];
      h8 hi = wpack[base + 1];
#pragma unroll
      for (int e = 0; e < 8; ++e) {
        bf[nt][ki][e]     = lo[e];
        bf[nt][ki][e + 8] = hi[e];
      }
    }
  }

  // ---- hoist per-lane gate metadata + biases (reused for all 4 rows) ----
  int   coq[4], xrq[4];
  float bI[4], bF[4], bG[4], bO[4];
#pragma unroll
  for (int q = 0; q < 4; ++q) {
    int lin = q * 32 + lane;
    coq[q] = lin >> 4;
    xrq[q] = lin & 15;
    bI[q]  = bc[coq[q]];
    bF[q]  = bc[coq[q] + 8];
    bG[q]  = bc[coq[q] + 16];
    bO[q]  = bc[coq[q] + 24];
  }

  asm volatile("s_wait_asynccnt 0x0" ::: "memory");
  __syncthreads();

  float* zw = &zbuf[wv * (NG * 16)];

#pragma unroll
  for (int j = 0; j < ROWS; ++j) {
    // ---- implicit GEMM for row y0+j: K = (tap pair) x 16 ch per WMMA ----
    v8f acc0 = {};
    v8f acc1 = {};
#pragma unroll
    for (int ki = 0; ki < 5; ++ki) {
      int tp0 = 2 * ki;
      int ky0 = tp0 / 3, kx0 = tp0 - 3 * ky0;
      const h8* p0 = (const h8*)
          &lds_in[(((j + ky0) * 130) + (x0 + m + kx0)) * 16 + hs * 8];
      h8 lo = *p0;
      h8 hi = {};
      if (ki < 4) {
        int tp1 = tp0 + 1;
        int ky1 = tp1 / 3, kx1 = tp1 - 3 * ky1;
        const h8* p1 = (const h8*)
            &lds_in[(((j + ky1) * 130) + (x0 + m + kx1)) * 16 + hs * 8];
        hi = *p1;
      }
      v16h a;
#pragma unroll
      for (int e = 0; e < 8; ++e) { a[e] = lo[e]; a[e + 8] = hi[e]; }
      acc0 = __builtin_amdgcn_wmma_f32_16x16x32_f16(
          false, a, false, bf[0][ki], (short)0, acc0, false, false);
      acc1 = __builtin_amdgcn_wmma_f32_16x16x32_f16(
          false, a, false, bf[1][ki], (short)0, acc1, false, false);
    }

    // ---- scatter z tile to wave-private LDS region ----
    // C/D layout: lane l, vgpr v -> M = v + 8*(l>>4), N = l&15.
#pragma unroll
    for (int v = 0; v < 8; ++v) {
      int xr = v + 8 * hs;
      zw[m * 16 + xr]        = acc0[v];   // gate channels 0..15 (i, f)
      zw[(m + 16) * 16 + xr] = acc1[v];   // gate channels 16..31 (g, o)
    }

    // ---- LSTM pointwise update for row y0+j (same-wave DS is in-order) ----
    const int y = y0 + j;
#pragma unroll
    for (int q = 0; q < 4; ++q) {
      int co = coq[q];
      int xr = xrq[q];
      int x  = x0 + xr;
      float zi = zw[(co     ) * 16 + xr] + bI[q];
      float zf = zw[(co +  8) * 16 + xr] + bF[q];
      float zg = zw[(co + 16) * 16 + xr] + bG[q];
      float zo = zw[(co + 24) * 16 + xr] + bO[q];
      float si = fast_sigmoid(zi);
      float sf = fast_sigmoid(zf);
      float so = fast_sigmoid(zo);
      int cidx = ((b * COUT + co) * Hh + y) * Ww + x;
      float cn = sf * c_buf[cidx] + si * fast_tanh(zg);
      float hn = so * fast_tanh(cn);
      c_buf[cidx] = cn;
      // f16 h state, channel-interleaved, interior of padded ring
      h_out[(((b * HP) + (y + 1)) * WP + (x + 1)) * COUT + co] = (_Float16)hn;
      out[(((b * COUT + co) * SEQ + t) * Hh + y) * Ww + x] = hn;
    }
  }
}

// ---------------------------------------------------------------------------
// Workspace layout (~16.3 MB):
//   [0, 10240)        packed f16 weights (padded to 16 KB)
//   hA (4,326,400 B)  f16 h state, padded halo ring
//   hB (4,326,400 B)  f16 h state, padded halo ring
//   c  (8,388,608 B)  fp32 cell state
// h/c state stays L2-resident (192 MB) across the 32 sequential step
// launches; HBM traffic is essentially X read (268 MB) + output write
// (268 MB) ~= 23 us at 23.3 TB/s.
// ---------------------------------------------------------------------------
extern "C" void kernel_launch(void* const* d_in, const int* in_sizes, int n_in,
                              void* d_out, int out_size, void* d_ws,
                              size_t ws_size, hipStream_t stream) {
  const float* X  = (const float*)d_in[0];
  const float* Wc = (const float*)d_in[1];
  const float* bc = (const float*)d_in[2];
  float* out = (float*)d_out;

  _Float16* wpack = (_Float16*)d_ws;
  _Float16* hA = (_Float16*)((char*)d_ws + 16384);
  _Float16* hB = (_Float16*)((char*)d_ws + 16384 + HBYTES);
  float*    cB = (float*)((char*)d_ws + 16384 + 2 * HBYTES);

  convlstm_pack_w<<<20, 256, 0, stream>>>(Wc, wpack);
  // zero hA, hB (halo ring + initial state) and c, contiguous region
  convlstm_zero_f32<<<2048, 256, 0, stream>>>(
      (float*)hA, (int)((2 * HBYTES + CBYTES) / 4));

  _Float16* hbuf[2] = {hA, hB};
  for (int t = 0; t < SEQ; ++t) {
    convlstm_step<<<dim3(Hh / ROWS, Bb), 256, 0, stream>>>(
        X, (const h8*)wpack, bc, hbuf[t & 1], hbuf[(t + 1) & 1], cB, out, t);
  }
}
